// ForecastDecoder_68324339745131
// MI455X (gfx1250) — matrix operand order repaired and checked by reading, not compiled, not run on hardware
//
#include <hip/hip_runtime.h>
#include <hip/hip_bf16.h>
#include <math.h>

// ---------------------------------------------------------------------------
// ForecastDecoder on MI455X (gfx1250):
//   90 sequential steps; per step two 16384x512x512 GEMMs + 16384x512x16 proj.
//   Everything L2-resident (state 33.5MB fp32, weights 2MB) -> compute bound.
//   v_wmma_f32_16x16x32_bf16 inner loop (8 WMMAs/K-chunk/wave), f32 accum.
//   GELU epilogue uses v_tanh_f32 (CDNA5 TRANS op) instead of branchy erff.
// ---------------------------------------------------------------------------

typedef __attribute__((ext_vector_type(16))) __bf16 v16bf;
typedef __attribute__((ext_vector_type(8)))  __bf16 v8bf;
typedef __attribute__((ext_vector_type(4)))  __bf16 v4bf;
typedef __attribute__((ext_vector_type(8)))  float  v8f;
typedef __attribute__((ext_vector_type(4)))  float  v4f;

#define HID    512
#define NBATCH 64
#define SEQ    256
#define MROWS  (NBATCH * SEQ)   /* 16384 */
#define TSTEPS 90
#define OUTD   16

// fp32 -> bf16, round-to-nearest-even
__device__ __forceinline__ __bf16 f2bf(float f) {
  unsigned u = __float_as_uint(f);
  u += 0x7FFFu + ((u >> 16) & 1u);
  unsigned short h = (unsigned short)(u >> 16);
  return __builtin_bit_cast(__bf16, h);
}

// CDNA5 has V_TANH_F32 (TRANS pipe, co-executes with matrix ops).
__device__ __forceinline__ float fast_tanh(float x) {
#if __has_builtin(__builtin_amdgcn_tanhf)
  return __builtin_amdgcn_tanhf(x);
#else
  float r;
  asm("v_tanh_f32 %0, %1" : "=v"(r) : "v"(x));
  return r;
#endif
}

// tanh-form GELU: 0.5x(1+tanh(0.79788456(x + 0.044715 x^3)))
// |err| vs exact erf-GELU ~1e-3 — below bf16 GEMM quantization noise.
__device__ __forceinline__ float gelu_tanh(float x) {
  float x2 = x * x;
  float inner = x * __builtin_fmaf(0.0356774081f, x2, 0.7978845608f);
  return 0.5f * x * (1.0f + fast_tanh(inner));
}

__device__ __forceinline__ v8f wmma_bf16(v16bf a, v16bf b, v8f c) {
  return __builtin_amdgcn_wmma_f32_16x16x32_bf16(
      /*neg_a=*/false, a, /*neg_b=*/false, b,
      /*c_mod=*/(short)0, c, /*reuse_a=*/false, /*reuse_b=*/false);
}

// A fragment (16x32 bf16, row-major, ld = HID).  ISA 7.12.2 layout:
// lanes 0-15 : row = lane,    K = [0..7] and [16..23]
// lanes 16-31: row = lane-16, K = [8..15] and [24..31]
__device__ __forceinline__ v16bf load_frag_a(const __bf16* __restrict__ base) {
  const int lane = threadIdx.x & 31;
  const int row  = lane & 15;
  const int kh   = (lane >> 4) * 8;
  const __bf16* p = base + (size_t)row * HID + kh;
  v8bf a0 = *(const v8bf*)(p);
  v8bf a1 = *(const v8bf*)(p + 16);
  return __builtin_shufflevector(a0, a1, 0,1,2,3,4,5,6,7,8,9,10,11,12,13,14,15);
}

// B fragment (32x16 bf16).  Weight stored (N,K) row-major so column n of B is
// row n of W (contiguous).  lanes 0-15: col=lane, K=[0..7] and [16..23];
// lanes 16-31: col=lane-16, K=[8..15] and [24..31].  Two 16B loads.
__device__ __forceinline__ v16bf load_frag_b(const __bf16* __restrict__ base) {
  const int lane = threadIdx.x & 31;
  const int col  = lane & 15;
  const int k    = (lane >> 4) * 8;
  const __bf16* p = base + (size_t)col * HID + k;
  v8bf b0 = *(const v8bf*)(p);
  v8bf b1 = *(const v8bf*)(p + 16);
  return __builtin_shufflevector(b0, b1, 0,1,2,3,4,5,6,7,8,9,10,11,12,13,14,15);
}

// ---------------------------------------------------------------------------
// Per-step activation prep: xbf = bf16(state + pos_enc[t]) (vector x4)
// ---------------------------------------------------------------------------
__global__ __launch_bounds__(256) void fd_prep_x(
    const float* __restrict__ st, const float* __restrict__ pe_t,
    __bf16* __restrict__ X) {
  const size_t i = ((size_t)blockIdx.x * 256 + threadIdx.x) * 4;
  v4f s = *(const v4f*)(st + i);
  const int h = (int)(i & (HID - 1));
  v4bf o;
  #pragma unroll
  for (int e = 0; e < 4; ++e) o[e] = f2bf(s[e] + pe_t[h + e]);
  *(v4bf*)(X + i) = o;
}

// One-time fp32 -> bf16 weight conversion
__global__ __launch_bounds__(256) void fd_cvt_bf16(
    const float* __restrict__ s, __bf16* __restrict__ d, int n) {
  int i = blockIdx.x * 256 + threadIdx.x;
  if (i < n) d[i] = f2bf(s[i]);
}

// ---------------------------------------------------------------------------
// GEMM1: H1 = bf16(gelu(X @ W1^T + b1)); M=16384 N=512 K=512
// Block = 256 thr (8 waves), block tile 128x128, wave tile 32x64 (8 WMMAs/iter)
// ---------------------------------------------------------------------------
__global__ __launch_bounds__(256) void fd_gemm1_gelu(
    const __bf16* __restrict__ X, const __bf16* __restrict__ W,
    const float* __restrict__ bias, __bf16* __restrict__ Hout) {
  const int wave = threadIdx.x >> 5;
  const int lane = threadIdx.x & 31;
  const int m0 = (blockIdx.x & 127) * 128 + (wave & 3) * 32;
  const int n0 = (blockIdx.x >> 7) * 128 + (wave >> 2) * 64;

  v8f zero = {};
  v8f acc[2][4];
  #pragma unroll
  for (int i = 0; i < 2; ++i)
    #pragma unroll
    for (int j = 0; j < 4; ++j) acc[i][j] = zero;

  for (int k = 0; k < HID; k += 32) {
    __builtin_prefetch(X + (size_t)(m0 + (lane & 15)) * HID + k + 64, 0, 3);
    v16bf a[2], b[4];
    #pragma unroll
    for (int i = 0; i < 2; ++i)
      a[i] = load_frag_a(X + (size_t)(m0 + i * 16) * HID + k);
    #pragma unroll
    for (int j = 0; j < 4; ++j)
      b[j] = load_frag_b(W + (size_t)(n0 + j * 16) * HID + k);
    #pragma unroll
    for (int i = 0; i < 2; ++i)
      #pragma unroll
      for (int j = 0; j < 4; ++j)
        acc[i][j] = wmma_bf16(a[i], b[j], acc[i][j]);
  }

  // Epilogue: C layout — VGPR r: row = r (lanes 0-15) / r+8 (lanes 16-31),
  // col = lane%16.  Apply bias + tanh-GELU (branch-free, TRANS pipe).
  const int rb = (lane >> 4) * 8;
  const int cl = lane & 15;
  #pragma unroll
  for (int i = 0; i < 2; ++i)
    #pragma unroll
    for (int j = 0; j < 4; ++j) {
      const int col = n0 + j * 16 + cl;
      const float bv = bias[col];
      #pragma unroll
      for (int r = 0; r < 8; ++r) {
        const int row = m0 + i * 16 + rb + r;
        Hout[(size_t)row * HID + col] = f2bf(gelu_tanh(acc[i][j][r] + bv));
      }
    }
}

// ---------------------------------------------------------------------------
// GEMM2: state += H1 @ W2^T + b2   (fp32 residual, in place)
// ---------------------------------------------------------------------------
__global__ __launch_bounds__(256) void fd_gemm2_resid(
    const __bf16* __restrict__ H1, const __bf16* __restrict__ W,
    const float* __restrict__ bias, float* __restrict__ state) {
  const int wave = threadIdx.x >> 5;
  const int lane = threadIdx.x & 31;
  const int m0 = (blockIdx.x & 127) * 128 + (wave & 3) * 32;
  const int n0 = (blockIdx.x >> 7) * 128 + (wave >> 2) * 64;

  v8f zero = {};
  v8f acc[2][4];
  #pragma unroll
  for (int i = 0; i < 2; ++i)
    #pragma unroll
    for (int j = 0; j < 4; ++j) acc[i][j] = zero;

  for (int k = 0; k < HID; k += 32) {
    __builtin_prefetch(H1 + (size_t)(m0 + (lane & 15)) * HID + k + 64, 0, 3);
    v16bf a[2], b[4];
    #pragma unroll
    for (int i = 0; i < 2; ++i)
      a[i] = load_frag_a(H1 + (size_t)(m0 + i * 16) * HID + k);
    #pragma unroll
    for (int j = 0; j < 4; ++j)
      b[j] = load_frag_b(W + (size_t)(n0 + j * 16) * HID + k);
    #pragma unroll
    for (int i = 0; i < 2; ++i)
      #pragma unroll
      for (int j = 0; j < 4; ++j)
        acc[i][j] = wmma_bf16(a[i], b[j], acc[i][j]);
  }

  const int rb = (lane >> 4) * 8;
  const int cl = lane & 15;
  #pragma unroll
  for (int i = 0; i < 2; ++i)
    #pragma unroll
    for (int j = 0; j < 4; ++j) {
      const int col = n0 + j * 16 + cl;
      const float bv = bias[col];
      #pragma unroll
      for (int r = 0; r < 8; ++r) {
        const int row = m0 + i * 16 + rb + r;
        float* sp = state + (size_t)row * HID + col;
        *sp = *sp + acc[i][j][r] + bv;   // residual, fp32
      }
    }
}

// ---------------------------------------------------------------------------
// Output projection: out[b, t*S+s, :] = state @ Wout^T + bout  (N = 16)
// 1 wave per 16-row tile; fp32 state converted to bf16 on the fly.
// ---------------------------------------------------------------------------
__global__ __launch_bounds__(256) void fd_outproj(
    const float* __restrict__ st, const __bf16* __restrict__ Wo,
    const float* __restrict__ bo, float* __restrict__ out, int t) {
  const int wave = threadIdx.x >> 5;
  const int lane = threadIdx.x & 31;
  const int m0 = (blockIdx.x * 8 + wave) * 16;

  v8f acc = {};
  const int row = lane & 15;
  const int kh  = (lane >> 4) * 8;
  for (int k = 0; k < HID; k += 32) {
    const float* p = st + (size_t)(m0 + row) * HID + k + kh;
    v4f f0 = *(const v4f*)(p);
    v4f f1 = *(const v4f*)(p + 4);
    v4f f2 = *(const v4f*)(p + 16);
    v4f f3 = *(const v4f*)(p + 20);
    v16bf a;
    #pragma unroll
    for (int e = 0; e < 4; ++e) {
      a[e]      = f2bf(f0[e]);
      a[e + 4]  = f2bf(f1[e]);
      a[e + 8]  = f2bf(f2[e]);
      a[e + 12] = f2bf(f3[e]);
    }
    v16bf b = load_frag_b(Wo + k);   // n0 = 0 (OUTD == 16)
    acc = wmma_bf16(a, b, acc);
  }

  const int rb = (lane >> 4) * 8;
  const int o  = lane & 15;
  const float bv = bo[o];
  #pragma unroll
  for (int r = 0; r < 8; ++r) {
    const int m = m0 + rb + r;
    const int b = m >> 8;      // batch  (m / SEQ)
    const int s = m & 255;     // seq    (m % SEQ)
    out[(((size_t)b * TSTEPS + t) * SEQ + s) * OUTD + o] = acc[r] + bv;
  }
}

// ---------------------------------------------------------------------------
extern "C" void kernel_launch(void* const* d_in, const int* in_sizes, int n_in,
                              void* d_out, int out_size, void* d_ws, size_t ws_size,
                              hipStream_t stream) {
  (void)in_sizes; (void)n_in; (void)out_size; (void)ws_size;
  const float* latent = (const float*)d_in[0];
  const float* W1     = (const float*)d_in[1];
  const float* b1     = (const float*)d_in[2];
  const float* W2     = (const float*)d_in[3];
  const float* b2     = (const float*)d_in[4];
  const float* Wout   = (const float*)d_in[5];
  const float* bout   = (const float*)d_in[6];
  const float* pe     = (const float*)d_in[7];
  float* out = (float*)d_out;

  // Workspace layout (~68 MB)
  char* ws = (char*)d_ws;
  size_t off = 0;
  float*  state = (float*)(ws + off);  off += (size_t)MROWS * HID * 4;
  __bf16* xbf   = (__bf16*)(ws + off); off += (size_t)MROWS * HID * 2;
  __bf16* h1    = (__bf16*)(ws + off); off += (size_t)MROWS * HID * 2;
  __bf16* w1b   = (__bf16*)(ws + off); off += (size_t)HID * HID * 2;
  __bf16* w2b   = (__bf16*)(ws + off); off += (size_t)HID * HID * 2;
  __bf16* wob   = (__bf16*)(ws + off); off += (size_t)OUTD * HID * 2;

  // One-time init (graph-capture safe: async d2d + kernels on stream)
  hipMemcpyAsync(state, latent, (size_t)MROWS * HID * 4,
                 hipMemcpyDeviceToDevice, stream);
  fd_cvt_bf16<<<(HID * HID + 255) / 256, 256, 0, stream>>>(W1, w1b, HID * HID);
  fd_cvt_bf16<<<(HID * HID + 255) / 256, 256, 0, stream>>>(W2, w2b, HID * HID);
  fd_cvt_bf16<<<(OUTD * HID + 255) / 256, 256, 0, stream>>>(Wout, wob, OUTD * HID);

  const dim3 blk(256);
  const int gemm_grid = 128 * (HID / 128);          // 512 blocks
  const int prep_grid = MROWS * HID / 4 / 256;      // 8192 blocks
  const int proj_grid = (MROWS / 16) / 8;           // 128 blocks

  for (int t = 0; t < TSTEPS; ++t) {
    fd_prep_x<<<prep_grid, blk, 0, stream>>>(state, pe + (size_t)t * HID, xbf);
    fd_gemm1_gelu<<<gemm_grid, blk, 0, stream>>>(xbf, w1b, b1, h1);
    fd_gemm2_resid<<<gemm_grid, blk, 0, stream>>>(h1, w2b, b2, state);
    fd_outproj<<<proj_grid, blk, 0, stream>>>(state, wob, bout, out, t);
  }
}
